// ScaledDotProductAttention_54503134986660
// MI455X (gfx1250) — matrix operand order repaired and checked by reading, not compiled, not run on hardware
//
#include <hip/hip_runtime.h>

typedef __attribute__((ext_vector_type(16))) _Float16 v16h;
typedef __attribute__((ext_vector_type(8)))  _Float16 v8h;
typedef __attribute__((ext_vector_type(8)))  float    v8f;
typedef __attribute__((ext_vector_type(4)))  float    v4f;
typedef __attribute__((ext_vector_type(8)))  int      v8i;

#define BB 2
#define HH 16
#define LL 2048
#define DKK 64
#define DVV 64

// xor-16 lane exchange (wave32): ds_swizzle SWAPX16 (xor=0x10, and=0x1f)
__device__ __forceinline__ float swz16(float x) {
  return __int_as_float(__builtin_amdgcn_ds_swizzle(__float_as_int(x), 0x401f));
}
__device__ __forceinline__ int swz16i(int x) {
  return __builtin_amdgcn_ds_swizzle(x, 0x401f);
}
// pack two floats into a half2 dword (compiler emits v_cvt_pk_f16_f32)
__device__ __forceinline__ int pkh2(float a, float b) {
  union { _Float16 h[2]; int i; } u;
  u.h[0] = (_Float16)a; u.h[1] = (_Float16)b;
  return u.i;
}

// Assemble a 16x32 f16 A-operand from LDS: halves 0-7 at base, halves 8-15 at base+8.
__device__ __forceinline__ v16h ldsA(const _Float16* base) {
  v8h a = *(const v8h*)(base);
  v8h b = *(const v8h*)(base + 8);
  v16h d;
#pragma unroll
  for (int i = 0; i < 8; ++i) { d[i] = a[i]; d[8 + i] = b[i]; }
  return d;
}

// Load a 32x16 f16 B-operand from 16 consecutive fp32 (with scale).
__device__ __forceinline__ v16h loadB16(const float* p, float scale) {
  v4f a0 = *(const v4f*)(p);
  v4f a1 = *(const v4f*)(p + 4);
  v4f a2 = *(const v4f*)(p + 8);
  v4f a3 = *(const v4f*)(p + 12);
  v16h d;
#pragma unroll
  for (int i = 0; i < 4; ++i) {
    d[i]      = (_Float16)(a0[i] * scale);
    d[4 + i]  = (_Float16)(a1[i] * scale);
    d[8 + i]  = (_Float16)(a2[i] * scale);
    d[12 + i] = (_Float16)(a3[i] * scale);
  }
  return d;
}

__global__ __launch_bounds__(256) void flash_attn_causal(
    const float* __restrict__ Q, const float* __restrict__ K,
    const float* __restrict__ V, float* __restrict__ O) {
  // K tile staged row-major f16 [32 keys][64 feats]; V staged pre-transposed
  // f16 [64 vcols][32 keys]. Double buffered -> one barrier per 32-key step.
  __shared__ _Float16 Ksh[2][32 * 64];
  __shared__ _Float16 VTsh[2][64 * 32];

  const int t      = threadIdx.x;
  const int lane   = t & 31;
  const int wave   = t >> 5;
  const int bh     = blockIdx.x >> 4;          // 16 blocks of 128 queries per (b,h)
  const int q0blk  = (blockIdx.x & 15) << 7;   // first query row of this block
  const int q0     = q0blk + (wave << 4);      // this wave's 16-query tile
  const int lo     = lane & 15;
  const bool hiH   = lane >= 16;

  const float* Qb = Q + (size_t)bh * LL * DKK;
  const float* Kb = K + (size_t)bh * LL * DKK;
  const float* Vb = V + (size_t)bh * LL * DVV;
  float*       Ob = O + (size_t)bh * LL * DVV;

  // Q^T as B operands; fold 1/sqrt(dk) * log2(e) so softmax uses exp2 directly.
  const float SC = 0.125f * 1.4426950408889634f;
  const float* qrow = Qb + (size_t)(q0 + lo) * DKK + (hiH ? 16 : 0);
  const v16h bQ0 = loadB16(qrow, SC);        // feats 0..31
  const v16h bQ1 = loadB16(qrow + 32, SC);   // feats 32..63

  v8f oacc0 = {}, oacc1 = {}, oacc2 = {}, oacc3 = {};
  float mI = -__builtin_inff();
  float lI = 0.0f;

  const int qg   = q0 + lo;          // this lane's query index
  const int kend = q0 + 16;          // keys [0, kend) needed by this wave
  const int kmax = q0blk + 128;      // keys needed by the whole block (causal)

  // staging map: thread t handles key-row (t>>3), 8 feats/vcols at (t&7)*8
  const int srow = t >> 3;
  const int scol = (t & 7) << 3;

  // ---- software-pipelined staging: preload tile j=0 into registers ----
  v4f k0r, k1r, w0r, w1r;
  {
    const float* kp = Kb + (size_t)srow * DKK + scol;
    k0r = *(const v4f*)kp;  k1r = *(const v4f*)(kp + 4);
    const float* vp = Vb + (size_t)srow * DVV + scol;
    w0r = *(const v4f*)vp;  w1r = *(const v4f*)(vp + 4);
  }

  for (int j = 0; j < kmax; j += 32) {
    const int buf = (j >> 5) & 1;

    // ---- commit staged registers: K row-major f16, V transposed f16 ----
    {
      v8h kh;
#pragma unroll
      for (int i = 0; i < 4; ++i) { kh[i] = (_Float16)k0r[i]; kh[4 + i] = (_Float16)k1r[i]; }
      *(v8h*)&Ksh[buf][srow * 64 + scol] = kh;
#pragma unroll
      for (int i = 0; i < 4; ++i) {
        VTsh[buf][(scol + i)     * 32 + srow] = (_Float16)w0r[i];
        VTsh[buf][(scol + 4 + i) * 32 + srow] = (_Float16)w1r[i];
      }
    }
    __syncthreads();

    // ---- prefetch next tile (block-uniform); hides under WMMA/softmax ----
    if (j + 32 < kmax) {
      const float* kp = Kb + (size_t)(j + 32 + srow) * DKK + scol;
      k0r = *(const v4f*)kp;  k1r = *(const v4f*)(kp + 4);
      const float* vp = Vb + (size_t)(j + 32 + srow) * DVV + scol;
      w0r = *(const v4f*)vp;  w1r = *(const v4f*)(vp + 4);
    }

    if (j < kend) {   // wave-uniform guard: EXEC all-ones inside (WMMA-legal)
      const int cb = hiH ? 8 : 0;

      // ---- S^T tiles: c0 = keys j..j+15, c1 = keys j+16..j+31 (x queries) ----
      const _Float16* krow = &Ksh[buf][lo * 64 + (hiH ? 16 : 0)];
      v16h aK0 = ldsA(krow);
      v16h aK1 = ldsA(krow + 32);
      const _Float16* krow2 = krow + 16 * 64;
      v16h aK2 = ldsA(krow2);
      v16h aK3 = ldsA(krow2 + 32);
      v8f c0 = {}, c1 = {};
      c0 = __builtin_amdgcn_wmma_f32_16x16x32_f16(false, aK0, false, bQ0, (short)0, c0, false, false);
      c1 = __builtin_amdgcn_wmma_f32_16x16x32_f16(false, aK2, false, bQ0, (short)0, c1, false, false);
      c0 = __builtin_amdgcn_wmma_f32_16x16x32_f16(false, aK1, false, bQ1, (short)0, c0, false, false);
      c1 = __builtin_amdgcn_wmma_f32_16x16x32_f16(false, aK3, false, bQ1, (short)0, c1, false, false);

      // ---- causal mask: only the LAST active step can straddle the diagonal ----
      float s0[8], s1[8];
      if (j + 32 < kend) {              // fast path: all 32 keys <= q0 <= every query
#pragma unroll
        for (int r = 0; r < 8; ++r) { s0[r] = c0[r]; s1[r] = c1[r]; }
      } else {                          // masked tail (exactly one step per wave)
#pragma unroll
        for (int r = 0; r < 8; ++r) {
          int kk = j + r + cb;
          s0[r] = (kk      <= qg) ? c0[r] : -__builtin_inff();
          s1[r] = (kk + 16 <= qg) ? c1[r] : -__builtin_inff();
        }
      }

      // ---- online softmax (scores already in log2 domain) ----
      float bm = -__builtin_inff();
#pragma unroll
      for (int r = 0; r < 8; ++r) bm = fmaxf(bm, fmaxf(s0[r], s1[r]));
      bm = fmaxf(bm, swz16(bm));             // full 32-key max for this query
      const float mNew = fmaxf(mI, bm);
      const float fac  = __builtin_amdgcn_exp2f(mI - mNew);

      float p0[8], p1[8];
      float sum = 0.0f;
#pragma unroll
      for (int r = 0; r < 8; ++r) {
        p0[r] = __builtin_amdgcn_exp2f(s0[r] - mNew);
        p1[r] = __builtin_amdgcn_exp2f(s1[r] - mNew);
        sum += p0[r] + p1[r];
      }
      sum += swz16(sum);                     // full 32-key row sum
      lI = lI * fac + sum;
      mI = mNew;
#pragma unroll
      for (int r = 0; r < 8; ++r) {
        oacc0[r] *= fac; oacc1[r] *= fac; oacc2[r] *= fac; oacc3[r] *= fac;
      }

      // ---- pack P^T into a 32x16 B operand: pack half2 first, swizzle dwords ----
      int pk0[4], pk1[4];
#pragma unroll
      for (int i = 0; i < 4; ++i) {
        pk0[i] = pkh2(p0[2 * i], p0[2 * i + 1]);
        pk1[i] = pkh2(p1[2 * i], p1[2 * i + 1]);
      }
      v8i bPd;
#pragma unroll
      for (int i = 0; i < 4; ++i) {
        int pk0s = swz16i(pk0[i]);
        int pk1s = swz16i(pk1[i]);
        bPd[i]     = hiH ? pk1s : pk0[i];    // halves 0-7 : keys h / 16+h
        bPd[4 + i] = hiH ? pk1[i] : pk0s;    // halves 8-15: keys 8+h / 24+h
      }
      const v16h bP = __builtin_bit_cast(v16h, bPd);

      // ---- O^T += V^T(chunk) x P : 4 chunks of 16 V-columns (from LDS) ----
      const _Float16* vrow = &VTsh[buf][lo * 32 + (hiH ? 16 : 0)];
      v16h aV0 = ldsA(vrow);
      v16h aV1 = ldsA(vrow + 16 * 32);
      v16h aV2 = ldsA(vrow + 32 * 32);
      v16h aV3 = ldsA(vrow + 48 * 32);
      oacc0 = __builtin_amdgcn_wmma_f32_16x16x32_f16(false, aV0, false, bP, (short)0, oacc0, false, false);
      oacc1 = __builtin_amdgcn_wmma_f32_16x16x32_f16(false, aV1, false, bP, (short)0, oacc1, false, false);
      oacc2 = __builtin_amdgcn_wmma_f32_16x16x32_f16(false, aV2, false, bP, (short)0, oacc2, false, false);
      oacc3 = __builtin_amdgcn_wmma_f32_16x16x32_f16(false, aV3, false, bP, (short)0, oacc3, false, false);
    }
  }

  // ---- epilogue: O(row = q0+lo, col = chunk*16 + r + (hiH?8:0)) ----
  // Each chunk's 8 results are 8 consecutive, 16B-aligned floats -> b128 stores.
  const float inv = 1.0f / lI;
  float* orow = Ob + (size_t)(q0 + lo) * DVV + (hiH ? 8 : 0);
  const v8f* oa[4] = { &oacc0, &oacc1, &oacc2, &oacc3 };
#pragma unroll
  for (int c = 0; c < 4; ++c) {
    const v8f& a = *oa[c];
    v4f lo4, hi4;
#pragma unroll
    for (int i = 0; i < 4; ++i) { lo4[i] = a[i] * inv; hi4[i] = a[4 + i] * inv; }
    *(v4f*)(orow + 16 * c)     = lo4;
    *(v4f*)(orow + 16 * c + 4) = hi4;
  }
}

extern "C" void kernel_launch(void* const* d_in, const int* in_sizes, int n_in,
                              void* d_out, int out_size, void* d_ws, size_t ws_size,
                              hipStream_t stream) {
  const float* Q = (const float*)d_in[0];
  const float* K = (const float*)d_in[1];
  const float* V = (const float*)d_in[2];
  // d_in[3] is the boolean causal mask; causality is hardcoded in the kernel.
  float* O = (float*)d_out;

  const int blocks = BB * HH * (LL / 128);   // 512 blocks, 128 queries each
  flash_attn_causal<<<blocks, 256, 0, stream>>>(Q, K, V, O);
}
